// MultiHeadAttention_50594714747117
// MI455X (gfx1250) — matrix operand order repaired and checked by reading, not compiled, not run on hardware
//
#include <hip/hip_runtime.h>

// Problem constants (B=4, T=2048, D=1024, H=16, dh=64)
#define BB 4
#define TT 2048
#define DD 1024
#define HH 16
#define DHH 64
#define D3 3072
#define MTOT 8192   // B*T

typedef __attribute__((ext_vector_type(16))) __bf16 v16bf;
typedef __attribute__((ext_vector_type(8)))  float  v8f;

union BF16Frag { v16bf v; unsigned int u[8]; };

// 16-byte async global->LDS copy (GLOBAL_LOAD_ASYNC_TO_LDS_B128, ASYNCcnt).
// GV addressing: vdst = per-lane LDS byte offset, vaddr = 64-bit global addr.
// Low 32 bits of a generic pointer to LDS are the LDS byte address (ISA 10.2).
__device__ __forceinline__ void async_copy16(const void* gsrc, void* ldst) {
  unsigned lds_off = (unsigned)(unsigned long long)ldst;
  asm volatile("global_load_async_to_lds_b128 %0, %1, off"
               :: "v"(lds_off), "v"(gsrc) : "memory");
}

__device__ __forceinline__ void wait_async0() {
#if __has_builtin(__builtin_amdgcn_s_wait_asynccnt)
  __builtin_amdgcn_s_wait_asynccnt(0);
#else
  asm volatile("s_wait_asynccnt 0x0" ::: "memory");
#endif
}

__device__ __forceinline__ unsigned short f32_to_bf16(float f) {
  union { float f; unsigned int u; } c; c.f = f;
  unsigned int x = c.u;
  // round-to-nearest-even; +/-inf maps to bf16 +/-inf
  return (unsigned short)((x + 0x7FFFu + ((x >> 16) & 1u)) >> 16);
}

__global__ __launch_bounds__(256) void cvt_bf16_kernel(const float* __restrict__ in,
                                                       unsigned short* __restrict__ out,
                                                       int n) {
  int i = blockIdx.x * 256 + threadIdx.x;
  if (i < n) out[i] = f32_to_bf16(in[i]);
}

// ---------------------------------------------------------------------------
// Tiled bf16 WMMA GEMM: C[M=8192, N] = A[8192,1024] * Bm[1024,N] + bias
// 128x128 tile per block, 256 threads = 8 waves (2 in M x 4 in N),
// each wave owns a 64x32 sub-tile = 4x2 grid of 16x16 WMMA accumulators.
// Double-buffered LDS: A staged by async global->LDS DMA, B staged through
// registers (transpose to K-major) with stores deferred past the WMMA block.
// ---------------------------------------------------------------------------
template <bool OUT_F32>
__global__ __launch_bounds__(256) void gemm_bf16_kernel(
    const unsigned short* __restrict__ A,    // bf16 [8192][1024]
    const unsigned short* __restrict__ Bm,   // bf16 [1024][N]
    const float* __restrict__ bias,          // [N]
    void* __restrict__ Cout,                 // bf16 or f32 [8192][N]
    int N) {
  __shared__ __align__(16) unsigned short Alds[2][128][32];   // [m][k] 2x8KB
  __shared__ __align__(16) unsigned short Blds[2][128][32];   // [n][k] 2x8KB

  const int tid  = threadIdx.x;
  const int wave = tid >> 5;
  const int lane = tid & 31;
  const int lrow = lane & 15;
  const int khA  = (lane < 16) ? 0 : 8;    // A-frag K half select
  const int khB  = (lane < 16) ? 0 : 16;   // B-frag K half select
  const int m0 = blockIdx.x * 128;
  const int n0 = blockIdx.y * 128;
  const int wm = wave & 1;   // 0..1
  const int wn = wave >> 1;  // 0..3

  unsigned breg[8];

  // A tile: 128 rows x 64B = 512 x B128 DMA transfers, 2 per thread
  auto stageA = [&](int buf, int kk) {
    #pragma unroll
    for (int it = 0; it < 2; ++it) {
      int t = tid + (it << 8);
      int r = t >> 2, sg = (t & 3) << 3;   // ushort offset within row
      async_copy16(&A[(size_t)(m0 + r) * 1024 + kk + sg], &Alds[buf][r][sg]);
    }
  };
  // B tile: 2048 dwords, 8 per thread, global -> regs
  auto loadB = [&](int kk) {
    #pragma unroll
    for (int q = 0; q < 8; ++q) {
      int i = tid + (q << 8);
      int k = i >> 6, nc = (i & 63) << 1;
      breg[q] = *(const unsigned int*)&Bm[(size_t)(kk + k) * N + n0 + nc];
    }
  };
  // regs -> LDS transposed to [n][k] so fragment K-pairs are contiguous
  auto storeB = [&](int buf) {
    #pragma unroll
    for (int q = 0; q < 8; ++q) {
      int i = tid + (q << 8);
      int k = i >> 6, nc = (i & 63) << 1;
      Blds[buf][nc][k]     = (unsigned short)(breg[q] & 0xFFFFu);
      Blds[buf][nc + 1][k] = (unsigned short)(breg[q] >> 16);
    }
  };

  v8f acc[4][2] = {};

  // Prologue: stage tile 0 into buffer 0
  stageA(0, 0);
  loadB(0);
  storeB(0);
  wait_async0();
  __syncthreads();

  int cur = 0;
  for (int kt = 0; kt < 32; ++kt) {
    const int kk = kt << 5;
    // Kick off next tile's DMA + global loads before computing
    if (kt < 31) {
      stageA(cur ^ 1, kk + 32);
      loadB(kk + 32);
    }

    BF16Frag afr[4], bfr[2];
    #pragma unroll
    for (int ti = 0; ti < 4; ti++) {
      int m = wm * 64 + ti * 16 + lrow;
      #pragma unroll
      for (int j = 0; j < 8; j++) {
        int k = ((j >> 2) << 4) + khA + ((j & 3) << 1);
        afr[ti].u[j] = *(const unsigned int*)&Alds[cur][m][k];
      }
    }
    #pragma unroll
    for (int tj = 0; tj < 2; tj++) {
      int n = wn * 32 + tj * 16 + lrow;
      #pragma unroll
      for (int j = 0; j < 8; j++) {
        int k = ((j >> 2) << 4) + khA + ((j & 3) << 1);
        bfr[tj].u[j] = *(const unsigned int*)&Blds[cur][n][k];
      }
    }
    #pragma unroll
    for (int ti = 0; ti < 4; ti++)
      #pragma unroll
      for (int tj = 0; tj < 2; tj++)
        acc[ti][tj] = __builtin_amdgcn_wmma_f32_16x16x32_bf16(
            false, afr[ti].v, false, bfr[tj].v, (short)0, acc[ti][tj],
            false, false);

    // Complete next tile's staging after compute has been issued
    if (kt < 31) {
      storeB(cur ^ 1);
      wait_async0();
    }
    __syncthreads();
    cur ^= 1;
  }

  // Epilogue: C layout (VGPR r -> M=r / r+8 by lane half; N = lane&15)
  #pragma unroll
  for (int ti = 0; ti < 4; ti++) {
    #pragma unroll
    for (int tj = 0; tj < 2; tj++) {
      #pragma unroll
      for (int r = 0; r < 8; r++) {
        int m = m0 + wm * 64 + ti * 16 + r + ((lane < 16) ? 0 : 8);
        int n = n0 + wn * 32 + tj * 16 + lrow;
        float val = acc[ti][tj][r] + bias[n];
        if (OUT_F32)
          ((float*)Cout)[(size_t)m * N + n] = val;
        else
          ((unsigned short*)Cout)[(size_t)m * N + n] = f32_to_bf16(val);
      }
    }
  }
}

// ---------------------------------------------------------------------------
// Attention (faithful: causal mask to -inf, NO softmax):
//   S = Q K^T * 1/sqrt(64); masked -> -inf; O = S V
// One block per (qblk, h, b): 128 query rows. 8 waves x 16 query rows.
// kqv buffer layout: bf16 [8192][3072], columns [0,1024)=K, [1024,2048)=Q,
// [2048,3072)=V, with per-head slice h*64..h*64+63.
// K tile staged via async global->LDS DMA; V transposed through VGPRs.
// ---------------------------------------------------------------------------
__global__ __launch_bounds__(256) void attn_kernel(
    const unsigned short* __restrict__ kqvc,   // bf16 [8192][3072]
    unsigned short* __restrict__ attnb) {      // bf16 [8192][1024]
  __shared__ __align__(16) unsigned short Klds[128][64];     // [key][d] 16KB
  __shared__ __align__(16) unsigned short Vt[64][128];       // [c][key] 16KB
  __shared__ __align__(16) unsigned short Slds[8][16][32];   // per-wave 8KB

  const int qblk = blockIdx.x;   // 0..15
  const int h    = blockIdx.y;   // 0..15
  const int b    = blockIdx.z;   // 0..3

  const int tid  = threadIdx.x;
  const int wave = tid >> 5;
  const int lane = tid & 31;
  const int lrow = lane & 15;
  const int khA  = (lane < 16) ? 0 : 8;
  const int khB  = (lane < 16) ? 0 : 16;
  const int mh   = (lane < 16) ? 0 : 8;   // C-layout M offset by lane half

  const unsigned short* kq = kqvc + (size_t)(b * TT) * D3;
  const int q0 = qblk * 128 + wave * 16;   // first query row of this wave

  // Q fragments (16 rows x 64 d) loaded straight from global (K-contiguous)
  BF16Frag qfr[2];
  const unsigned short* qrow = kq + (size_t)(q0 + lrow) * D3 + DD + h * DHH;
  #pragma unroll
  for (int dc = 0; dc < 2; dc++)
    #pragma unroll
    for (int j = 0; j < 8; j++) {
      int k = dc * 32 + ((j >> 2) << 4) + khA + ((j & 3) << 1);
      qfr[dc].u[j] = *(const unsigned int*)&qrow[k];
    }

  v8f o[4] = {};
  const float scale = 0.125f;  // 1/sqrt(64)

  for (int jb = 0; jb <= qblk; jb++) {
    __syncthreads();
    const int kb0 = jb * 128;
    // Stage K tile [key][d] via async DMA: 128 rows x 128B = 1024 B128, 4/thr
    #pragma unroll
    for (int it = 0; it < 4; ++it) {
      int t = tid + (it << 8);
      int key = t >> 3, sg = (t & 7) << 3;
      async_copy16(&kq[(size_t)(kb0 + key) * D3 + h * DHH + sg],
                   &Klds[key][sg]);
    }
    // Stage V tile transposed [c][key] (overlaps the in-flight K DMA)
    #pragma unroll
    for (int i = tid; i < 128 * 32; i += 256) {
      int key = i >> 5, cc = (i & 31) << 1;
      unsigned int val = *(const unsigned int*)
          &kq[(size_t)(kb0 + key) * D3 + 2 * DD + h * DHH + cc];
      Vt[cc][key]     = (unsigned short)(val & 0xFFFFu);
      Vt[cc + 1][key] = (unsigned short)(val >> 16);
    }
    wait_async0();
    __syncthreads();

    const bool diag = (jb == qblk);
    for (int st = 0; st < 4; st++) {   // 32 keys per iteration
      // S(16q x 32k) in two 16x16 accumulators
      v8f s[2] = {};
      #pragma unroll
      for (int kt = 0; kt < 2; kt++) {
        int nkey = st * 32 + kt * 16 + lrow;
        #pragma unroll
        for (int dc = 0; dc < 2; dc++) {
          BF16Frag bfr;
          #pragma unroll
          for (int j = 0; j < 8; j++) {
            int k = dc * 32 + ((j >> 2) << 4) + khA + ((j & 3) << 1);
            bfr.u[j] = *(const unsigned int*)&Klds[nkey][k];
          }
          s[kt] = __builtin_amdgcn_wmma_f32_16x16x32_bf16(
              false, qfr[dc].v, false, bfr.v, (short)0, s[kt], false, false);
        }
      }
      // Scale, causal mask (-inf), re-layout S through per-wave LDS
      #pragma unroll
      for (int kt = 0; kt < 2; kt++) {
        #pragma unroll
        for (int r = 0; r < 8; r++) {
          int ml = r + mh;
          int qg = q0 + ml;
          int kg = kb0 + st * 32 + kt * 16 + lrow;
          float v = s[kt][r] * scale;
          if (diag && kg > qg) v = -__builtin_inff();
          Slds[wave][ml][kt * 16 + lrow] = f32_to_bf16(v);
        }
      }
      // A-fragment of S (16x32 over keys); same-wave DS ops are in-order
      BF16Frag sfr;
      #pragma unroll
      for (int j = 0; j < 8; j++) {
        int k = ((j >> 2) << 4) + khA + ((j & 3) << 1);
        sfr.u[j] = *(const unsigned int*)&Slds[wave][lrow][k];
      }
      // O += S @ V over the 32-key chunk, 4 column tiles of 16
      #pragma unroll
      for (int ct = 0; ct < 4; ct++) {
        BF16Frag vfr;
        int n = ct * 16 + lrow;
        #pragma unroll
        for (int j = 0; j < 8; j++) {
          int k = st * 32 + ((j >> 2) << 4) + khA + ((j & 3) << 1);
          vfr.u[j] = *(const unsigned int*)&Vt[n][k];
        }
        o[ct] = __builtin_amdgcn_wmma_f32_16x16x32_bf16(
            false, sfr.v, false, vfr.v, (short)0, o[ct], false, false);
      }
    }
  }

  // Store O as bf16 into [B*T][1024] at columns h*64..h*64+63
  #pragma unroll
  for (int ct = 0; ct < 4; ct++) {
    #pragma unroll
    for (int r = 0; r < 8; r++) {
      int t = q0 + r + mh;
      int c = ct * 16 + lrow;
      attnb[(size_t)(b * TT + t) * DD + h * DHH + c] = f32_to_bf16(o[ct][r]);
    }
  }
}

extern "C" void kernel_launch(void* const* d_in, const int* in_sizes, int n_in,
                              void* d_out, int out_size, void* d_ws, size_t ws_size,
                              hipStream_t stream) {
  (void)in_sizes; (void)n_in; (void)out_size; (void)ws_size;
  const float* x  = (const float*)d_in[0];   // [4,2048,1024]
  const float* w1 = (const float*)d_in[1];   // [1024,3072]
  const float* b1 = (const float*)d_in[2];   // [3072]
  const float* w2 = (const float*)d_in[3];   // [1024,1024]
  const float* b2 = (const float*)d_in[4];   // [1024]
  float* out = (float*)d_out;                // [4,2048,1024]

  char* ws = (char*)d_ws;
  unsigned short* xb    = (unsigned short*)(ws);                        // 16 MB
  unsigned short* w1b   = (unsigned short*)(ws + (size_t)16 * 1048576); // 6 MB
  unsigned short* w2b   = (unsigned short*)(ws + (size_t)22 * 1048576); // 2 MB
  unsigned short* kqvc  = (unsigned short*)(ws + (size_t)24 * 1048576); // 48 MB
  unsigned short* attnb = xb;  // xb dead after KQV GEMM; reuse its region

  // fp32 -> bf16 conversions
  cvt_bf16_kernel<<<(MTOT * DD + 255) / 256, 256, 0, stream>>>(x, xb, MTOT * DD);
  cvt_bf16_kernel<<<(DD * D3 + 255) / 256, 256, 0, stream>>>(w1, w1b, DD * D3);
  cvt_bf16_kernel<<<(DD * DD + 255) / 256, 256, 0, stream>>>(w2, w2b, DD * DD);

  // kqv = x @ w1 + b1  (bf16 out, [8192][3072])
  gemm_bf16_kernel<false><<<dim3(MTOT / 128, D3 / 128), 256, 0, stream>>>(
      xb, w1b, b1, (void*)kqvc, D3);

  // per-head causal attention (no softmax, faithful to reference)
  attn_kernel<<<dim3(TT / 128, HH, BB), 256, 0, stream>>>(kqvc, attnb);

  // out = attn_out @ w2 + b2  (f32 out)
  gemm_bf16_kernel<true><<<dim3(MTOT / 128, DD / 128), 256, 0, stream>>>(
      attnb, w2b, b2, (void*)out, DD);
}
